// Block_78683800863152
// MI455X (gfx1250) — matrix operand order repaired and checked by reading, not compiled, not run on hardware
//
#include <hip/hip_runtime.h>
#include <hip/hip_bf16.h>

// ---------------------------------------------------------------------------
// Mamba block forward for MI455X (gfx1250, wave32, WMMA).
// GEMMs run as bf16 WMMA (v_wmma_f32_16x16x32_bf16) with f32 accumulation.
// Weights fit in the 192MB L2 so fragments are loaded straight from global.
// All NT B-fragments are loaded before the first WMMA of each K-step so the
// scheduler can overlap matrix ops with outstanding loads (partial loadcnt
// waits instead of s_wait_loadcnt 0).
// ---------------------------------------------------------------------------

typedef __bf16 bf16_t;
typedef __attribute__((ext_vector_type(16))) __bf16 v16bf;
typedef __attribute__((ext_vector_type(8)))  __bf16 v8bf;
typedef __attribute__((ext_vector_type(8)))  float  v8f;

#define D_MODEL  1024
#define D_STATE  16
#define D_CONV   4
#define D_INNER  2048
#define DT_RANK  64
#define XDBL_W   (DT_RANK + 2 * D_STATE)   // 96
#define SEQ_B    2
#define SEQ_L    1024
#define NROWS    (SEQ_B * SEQ_L)           // 2048
#define EPS_RMS  1e-5f

__device__ __forceinline__ float sigmoidf_(float x) { return 1.0f / (1.0f + __expf(-x)); }

// ---------------------------------------------------------------------------
// f32 -> bf16 cast (weights feeding WMMA)
// ---------------------------------------------------------------------------
__global__ void cast_f32_bf16(const float* __restrict__ in, bf16_t* __restrict__ out, int n) {
    int i = blockIdx.x * blockDim.x + threadIdx.x;
    if (i < n) out[i] = (bf16_t)in[i];
}

// ---------------------------------------------------------------------------
// RMSNorm (faithful: rms = ||x||/sqrt(d); x/(rms+eps)*w) + residual copy
// ---------------------------------------------------------------------------
__global__ void rmsnorm_kernel(const float* __restrict__ x, const float* __restrict__ w,
                               bf16_t* __restrict__ hB, float* __restrict__ resid) {
    __shared__ float red[256];
    int row = blockIdx.x;
    const float* xr = x + (size_t)row * D_MODEL;
    float ss = 0.0f;
    for (int i = threadIdx.x; i < D_MODEL; i += 256) {
        float v = xr[i];
        ss += v * v;
        resid[(size_t)row * D_MODEL + i] = v;
    }
    red[threadIdx.x] = ss;
    __syncthreads();
    for (int s = 128; s > 0; s >>= 1) {
        if (threadIdx.x < s) red[threadIdx.x] += red[threadIdx.x + s];
        __syncthreads();
    }
    float norm  = sqrtf(red[0]);
    float rms   = norm * 0.03125f;           // 1/sqrt(1024)
    float scale = 1.0f / (rms + EPS_RMS);
    for (int i = threadIdx.x; i < D_MODEL; i += 256)
        hB[(size_t)row * D_MODEL + i] = (bf16_t)(w[i] * xr[i] * scale);
}

// ---------------------------------------------------------------------------
// Wave-level NT GEMM: C[M,N](f32) = A[M,K](bf16,row-major) * B[N,K](bf16)^T
// Each wave owns a 16 x (16*NT) tile of C; K stepped by 32.
// Fragment layouts follow the CDNA5 16-bit WMMA VGPR tables:
//   A: lanes 0-15 row m, K {0..7,16..23}; lanes 16-31 row m, K {8..15,24..31}
//   B: lane (n%16) holds column n; lanes 0-15 K {0..7,16..23}; lanes 16-31 K {8..15,24..31}
//   C: element (m = r + 8*(lane>>4), n = lane&15) in vgpr r
// mode 0: plain store.  mode 1: softplus(c + bias[n]).
// Cb != nullptr: additionally store a bf16 copy of C (same layout, ldc).
// ---------------------------------------------------------------------------
template <int NT>
__global__ __launch_bounds__(256)
void gemm_nt_bf16(const bf16_t* __restrict__ A, const bf16_t* __restrict__ B,
                  float* __restrict__ C, bf16_t* __restrict__ Cb,
                  int M, int N, int K,
                  int lda, int ldb, int ldc,
                  const float* __restrict__ bias, int mode) {
    int wave = (blockIdx.x * blockDim.x + threadIdx.x) >> 5;
    int lane = threadIdx.x & 31;
    int tilesN = N / (16 * NT);
    int tm  = (wave / tilesN) * 16;
    int tn0 = (wave % tilesN) * 16 * NT;
    if (tm >= M) return;

    int lm    = lane & 15;
    int khalf = lane >> 4;

    v8f zero = {};
    v8f acc[NT];
#pragma unroll
    for (int t = 0; t < NT; ++t) acc[t] = zero;

    const bf16_t* arow = A + (size_t)(tm + lm) * lda + khalf * 8;
    const bf16_t* brow0 = B + (size_t)(tn0 + lm) * ldb + khalf * 8;

    for (int kk = 0; kk < K; kk += 32) {
        // ---- issue ALL loads for this K-step first --------------------
        v8bf alo = *(const v8bf*)(arow + kk);
        v8bf ahi = *(const v8bf*)(arow + kk + 16);

        v8bf blo[NT], bhi[NT];
#pragma unroll
        for (int t = 0; t < NT; ++t) {
            const bf16_t* brow = brow0 + (size_t)(t * 16) * ldb + kk;
            blo[t] = *(const v8bf*)(brow);
            bhi[t] = *(const v8bf*)(brow + 16);
        }

        __builtin_prefetch(arow + kk + 64, 0, 3);   // near-cache prefetch

        // ---- then consume with WMMAs ----------------------------------
        v16bf afrag;
#pragma unroll
        for (int i = 0; i < 8; ++i) { afrag[i] = alo[i]; afrag[i + 8] = ahi[i]; }

#pragma unroll
        for (int t = 0; t < NT; ++t) {
            v16bf bfrag;
#pragma unroll
            for (int i = 0; i < 8; ++i) { bfrag[i] = blo[t][i]; bfrag[i + 8] = bhi[t][i]; }
            acc[t] = __builtin_amdgcn_wmma_f32_16x16x32_bf16(
                false, afrag, false, bfrag, (short)0, acc[t], false, false);
        }
    }

    int mBase = tm + khalf * 8;
#pragma unroll
    for (int t = 0; t < NT; ++t) {
        int col = tn0 + t * 16 + lm;
#pragma unroll
        for (int r = 0; r < 8; ++r) {
            float v = acc[t][r];
            if (mode == 1) {                       // softplus(c + bias)
                float xv = v + bias[col];
                v = (xv > 20.0f) ? xv : log1pf(__expf(xv));
            }
            size_t o = (size_t)(mBase + r) * ldc + col;
            C[o] = v;
            if (Cb) Cb[o] = (bf16_t)v;
        }
    }
}

// ---------------------------------------------------------------------------
// Causal depthwise conv1d (taps=4) + SiLU. x lives in xz columns [0,D_INNER).
// Writes xc (f32, for scan) and xcB (bf16, for x_proj GEMM).
// ---------------------------------------------------------------------------
__global__ void conv_silu_kernel(const float* __restrict__ xz, const float* __restrict__ cw,
                                 const float* __restrict__ cb,
                                 float* __restrict__ xc, bf16_t* __restrict__ xcB) {
    int gid = blockIdx.x * blockDim.x + threadIdx.x;
    if (gid >= SEQ_B * SEQ_L * D_INNER) return;
    int d = gid % D_INNER;
    int l = (gid / D_INNER) % SEQ_L;
    int b = gid / (D_INNER * SEQ_L);

    float acc = cb[d];
#pragma unroll
    for (int k = 0; k < D_CONV; ++k) {
        int ls = l - (D_CONV - 1) + k;
        if (ls >= 0)
            acc += xz[(size_t)(b * SEQ_L + ls) * (2 * D_INNER) + d] * cw[d * D_CONV + k];
    }
    float v = acc * sigmoidf_(acc);
    size_t o = (size_t)(b * SEQ_L + l) * D_INNER + d;
    xc[o]  = v;
    xcB[o] = (bf16_t)v;
}

// ---------------------------------------------------------------------------
// Selective scan: one thread per (b,d). 16-wide state in registers,
// sequential over L. Fuses y += D*xc and the silu(z) gate; emits bf16 y.
// ---------------------------------------------------------------------------
__global__ void scan_kernel(const float* __restrict__ dt, const float* __restrict__ xc,
                            const float* __restrict__ xdbl, const float* __restrict__ xz,
                            const float* __restrict__ A_log, const float* __restrict__ Dp,
                            bf16_t* __restrict__ yB) {
    int gid = blockIdx.x * blockDim.x + threadIdx.x;
    if (gid >= SEQ_B * D_INNER) return;
    int d = gid % D_INNER;
    int b = gid / D_INNER;

    float Ad[D_STATE];
#pragma unroll
    for (int n = 0; n < D_STATE; ++n) Ad[n] = -__expf(A_log[d * D_STATE + n]);

    float st[D_STATE];
#pragma unroll
    for (int n = 0; n < D_STATE; ++n) st[n] = 0.0f;

    float Dd = Dp[d];

    for (int l = 0; l < SEQ_L; ++l) {
        size_t row = (size_t)(b * SEQ_L + l);
        float dtv = dt[row * D_INNER + d];
        float xcv = xc[row * D_INNER + d];
        const float* bm = xdbl + row * XDBL_W + DT_RANK;   // B_t  (16)
        const float* cm = bm + D_STATE;                    // C_t  (16)
        float dbu = dtv * xcv;
        float y = 0.0f;
#pragma unroll
        for (int n = 0; n < D_STATE; ++n) {
            float dA = __expf(dtv * Ad[n]);
            st[n] = dA * st[n] + dbu * bm[n];
            y += st[n] * cm[n];
        }
        y += Dd * xcv;
        float zv = xz[row * (2 * D_INNER) + D_INNER + d];
        y *= zv * sigmoidf_(zv);
        yB[row * D_INNER + d] = (bf16_t)y;
    }
}

// ---------------------------------------------------------------------------
// Host-side orchestration
// ---------------------------------------------------------------------------
extern "C" void kernel_launch(void* const* d_in, const int* in_sizes, int n_in,
                              void* d_out, int out_size, void* d_ws, size_t ws_size,
                              hipStream_t stream) {
    const float* hs        = (const float*)d_in[0];
    const float* norm_w    = (const float*)d_in[1];
    const float* in_proj_w = (const float*)d_in[2];   // (4096,1024)
    const float* conv_w    = (const float*)d_in[3];   // (2048,4)
    const float* conv_b    = (const float*)d_in[4];
    const float* x_proj_w  = (const float*)d_in[5];   // (96,2048)
    const float* dt_proj_w = (const float*)d_in[6];   // (2048,64)
    const float* dt_proj_b = (const float*)d_in[7];
    const float* A_log     = (const float*)d_in[8];   // (2048,16)
    const float* Dp        = (const float*)d_in[9];
    const float* out_proj_w= (const float*)d_in[10];  // (1024,2048)

    float* out   = (float*)d_out;                     // (B,L,D_MODEL)
    float* resid = out + (size_t)NROWS * D_MODEL;     // second output

    // workspace carve-up (256B aligned)
    char* ws = (char*)d_ws;
    size_t off = 0;
    auto alloc = [&](size_t bytes) -> void* {
        void* p = ws + off;
        off = (off + bytes + 255) & ~(size_t)255;
        return p;
    };
    bf16_t* hB     = (bf16_t*)alloc((size_t)NROWS * D_MODEL * 2);
    bf16_t* wInB   = (bf16_t*)alloc((size_t)2 * D_INNER * D_MODEL * 2);
    bf16_t* wXB    = (bf16_t*)alloc((size_t)XDBL_W * D_INNER * 2);
    bf16_t* wDtB   = (bf16_t*)alloc((size_t)D_INNER * DT_RANK * 2);
    bf16_t* wOutB  = (bf16_t*)alloc((size_t)D_MODEL * D_INNER * 2);
    float*  xz     = (float*) alloc((size_t)NROWS * 2 * D_INNER * 4);
    float*  xcf    = (float*) alloc((size_t)NROWS * D_INNER * 4);
    bf16_t* xcB    = (bf16_t*)alloc((size_t)NROWS * D_INNER * 2);
    float*  xdbl   = (float*) alloc((size_t)NROWS * XDBL_W * 4);
    bf16_t* xdblB  = (bf16_t*)alloc((size_t)NROWS * XDBL_W * 2);
    float*  dtbuf  = (float*) alloc((size_t)NROWS * D_INNER * 4);
    bf16_t* yB     = (bf16_t*)alloc((size_t)NROWS * D_INNER * 2);

    // 1) cast weights to bf16
    {
        int n;
        n = 2 * D_INNER * D_MODEL;
        cast_f32_bf16<<<(n + 255) / 256, 256, 0, stream>>>(in_proj_w, wInB, n);
        n = XDBL_W * D_INNER;
        cast_f32_bf16<<<(n + 255) / 256, 256, 0, stream>>>(x_proj_w, wXB, n);
        n = D_INNER * DT_RANK;
        cast_f32_bf16<<<(n + 255) / 256, 256, 0, stream>>>(dt_proj_w, wDtB, n);
        n = D_MODEL * D_INNER;
        cast_f32_bf16<<<(n + 255) / 256, 256, 0, stream>>>(out_proj_w, wOutB, n);
    }

    // 2) RMSNorm (+ residual output)
    rmsnorm_kernel<<<NROWS, 256, 0, stream>>>(hs, norm_w, hB, resid);

    // 3) in_proj: xz(2048x4096) = h(2048x1024) @ W^T   [NT=4 -> 8192 waves]
    {
        int waves = (NROWS / 16) * (2 * D_INNER / 64);
        gemm_nt_bf16<4><<<waves * 32 / 256, 256, 0, stream>>>(
            hB, wInB, xz, nullptr, NROWS, 2 * D_INNER, D_MODEL,
            D_MODEL, D_MODEL, 2 * D_INNER, nullptr, 0);
    }

    // 4) depthwise conv + SiLU
    {
        int n = SEQ_B * SEQ_L * D_INNER;
        conv_silu_kernel<<<(n + 255) / 256, 256, 0, stream>>>(xz, conv_w, conv_b, xcf, xcB);
    }

    // 5) x_proj: x_dbl(2048x96) = xc @ W^T  [NT=2 -> 3 N-tiles]; dual f32+bf16 store
    {
        int waves = (NROWS / 16) * (XDBL_W / 32);
        gemm_nt_bf16<2><<<(waves * 32 + 255) / 256, 256, 0, stream>>>(
            xcB, wXB, xdbl, xdblB, NROWS, XDBL_W, D_INNER,
            D_INNER, D_INNER, XDBL_W, nullptr, 0);
    }

    // 6) dt = softplus(dt_part @ dt_proj_w^T + b): K=64, lda=96 (slice of x_dbl)
    {
        int waves = (NROWS / 16) * (D_INNER / 64);
        gemm_nt_bf16<4><<<waves * 32 / 256, 256, 0, stream>>>(
            xdblB, wDtB, dtbuf, nullptr, NROWS, D_INNER, DT_RANK,
            XDBL_W, DT_RANK, D_INNER, dt_proj_b, 1);
    }

    // 7) selective scan (fused +D*xc and silu(z) gate) -> yB (bf16)
    {
        int n = SEQ_B * D_INNER;
        scan_kernel<<<(n + 255) / 256, 256, 0, stream>>>(
            dtbuf, xcf, xdbl, xz, A_log, Dp, yB);
    }

    // 8) out_proj: out(2048x1024) = y @ W^T
    {
        int waves = (NROWS / 16) * (D_MODEL / 64);
        gemm_nt_bf16<4><<<waves * 32 / 256, 256, 0, stream>>>(
            yB, wOutB, out, nullptr, NROWS, D_MODEL, D_INNER,
            D_INNER, D_INNER, D_MODEL, nullptr, 0);
    }
}
